// HybridQLSTM_65481071399042
// MI455X (gfx1250) — matrix operand order repaired and checked by reading, not compiled, not run on hardware
//
#include <hip/hip_runtime.h>
#include <hip/hip_bf16.h>
#include <math.h>

// ---------------- Problem constants ----------------
#define TT 512   // timesteps
#define BB 128   // batch
#define DD 512   // input dim
#define HH 512   // hidden dim
#define KDIM (DD + HH)          // 1024 (GEMM reduction dim)
#define NGATE 3                  // i, u(g), o  (forget gate via tiny MLP)
#define NOUT (NGATE * HH)        // 1536 columns of the fused gate GEMM
#define MTILES (BB / 16)         // 8
#define NTILES (NOUT / 16)       // 96
#define TOT_TILES (MTILES * NTILES) // 768
#define KSTEPS (KDIM / 32)       // 32 wmma k-steps

#define NBLK 64
#define TPB 256
#define WAVES_PER_BLK (TPB / 32)
#define NWAVES (NBLK * WAVES_PER_BLK)  // 512

// ---------------- Workspace layout (bytes) ----------------
#define OFF_SYNC 0                                   // 2 ints (counter, generation)
#define OFF_COMB 256                                 // f16 combined [128][1024] = x|h
#define OFF_C    (OFF_COMB + BB * KDIM * 2)          // c state f32 (C-layout order)
#define OFF_GI   (OFF_C  + BB * HH * 4)              // activated i gate (C-layout)
#define OFF_GU   (OFF_GI + BB * HH * 4)              // activated g (tanh)
#define OFF_GO   (OFF_GU + BB * HH * 4)              // activated o gate
#define OFF_F    (OFF_GO + BB * HH * 4)              // f[b], 128 f32 (pad to 1KB)
#define OFF_WT   (OFF_F  + 1024)                     // f16 weights [3][1024][512]
#define WT_ELEMS ((size_t)NGATE * KDIM * HH)

typedef __attribute__((ext_vector_type(16))) _Float16 v16h;
typedef __attribute__((ext_vector_type(8)))  _Float16 v8h;
typedef __attribute__((ext_vector_type(8)))  float    v8f;

// fast sigmoid: single v_exp_f32 + v_rcp_f32
__device__ __forceinline__ float fast_sigmoid(float x) {
    return __builtin_amdgcn_rcpf(1.0f + __expf(-x));
}
__device__ __forceinline__ float fast_tanh(float x) {
    return fmaf(2.0f, fast_sigmoid(2.0f * x), -1.0f);
}

// -------- init: zero state+sync, stage x_0 into combined, build f16 weights --------
__global__ void qlstm_init_convert(const float* __restrict__ X,
                                   const float* __restrict__ Wi,
                                   const float* __restrict__ Wu,
                                   const float* __restrict__ Wo,
                                   char* __restrict__ ws) {
    size_t idx = (size_t)blockIdx.x * blockDim.x + threadIdx.x;
    size_t nth = (size_t)gridDim.x * blockDim.x;

    // zero sync counters (first 256 bytes)
    float* z = (float*)ws;
    for (size_t i = idx; i < 64; i += nth) z[i] = 0.0f;

    // zero c state
    float* cbuf = (float*)(ws + OFF_C);
    for (size_t i = idx; i < (size_t)BB * HH; i += nth) cbuf[i] = 0.0f;

    // combined[b][j] : j<512 -> f16(x_0[b][j]) ; j>=512 -> h_0 = 0
    _Float16* comb = (_Float16*)(ws + OFF_COMB);
    for (size_t i = idx; i < (size_t)BB * KDIM; i += nth) {
        size_t b = i / KDIM, j = i % KDIM;
        comb[i] = (j < DD) ? (_Float16)X[b * DD + j] : (_Float16)0.0f;
    }

    // Wt[gate][k][n] = W_gate[n][k]   (f32 [H][K] row-major -> f16 k-major)
    _Float16* Wt = (_Float16*)(ws + OFF_WT);
    for (size_t i = idx; i < WT_ELEMS; i += nth) {
        size_t g = i / ((size_t)KDIM * HH);
        size_t r = i % ((size_t)KDIM * HH);
        size_t k = r / HH;
        size_t n = r % HH;
        const float* W = (g == 0) ? Wi : (g == 1) ? Wu : Wo;
        Wt[i] = (_Float16)W[n * KDIM + k];
    }
}

// -------- device-wide barrier (all NBLK blocks resident) --------
__device__ __forceinline__ void grid_barrier(int* cnt, int* gen) {
    __syncthreads();
    if (threadIdx.x == 0) {
        __threadfence();   // release prior global writes device-wide
        int g = __hip_atomic_load(gen, __ATOMIC_ACQUIRE, __HIP_MEMORY_SCOPE_AGENT);
        int a = __hip_atomic_fetch_add(cnt, 1, __ATOMIC_ACQ_REL, __HIP_MEMORY_SCOPE_AGENT);
        if (a == (int)gridDim.x - 1) {
            __hip_atomic_store(cnt, 0, __ATOMIC_RELAXED, __HIP_MEMORY_SCOPE_AGENT);
            __hip_atomic_fetch_add(gen, 1, __ATOMIC_RELEASE, __HIP_MEMORY_SCOPE_AGENT);
        } else {
            while (__hip_atomic_load(gen, __ATOMIC_ACQUIRE, __HIP_MEMORY_SCOPE_AGENT) == g) {
                __builtin_amdgcn_s_sleep(2);
            }
        }
        __threadfence();
    }
    __syncthreads();
}

// -------- persistent recurrent kernel --------
__global__ void qlstm_persistent(const float* __restrict__ X,    // [T][B][D]
                                 const float* __restrict__ Wf,   // [2][1024]
                                 const float* __restrict__ bfv,  // [2]
                                 const float* __restrict__ bi,   // [H]
                                 const float* __restrict__ bu,   // [H]
                                 const float* __restrict__ bo,   // [H]
                                 const float* __restrict__ Ws1,  // [4][2]
                                 const float* __restrict__ bs1,  // [4]
                                 const float* __restrict__ Ws2,  // [2][4]
                                 const float* __restrict__ bs2,  // [2]
                                 char* __restrict__ ws,
                                 float* __restrict__ out) {      // [T][B][H] + h + c
    const int tid  = threadIdx.x;
    const int lane = tid & 31;
    const int gw   = blockIdx.x * WAVES_PER_BLK + (tid >> 5);

    _Float16* comb = (_Float16*)(ws + OFF_COMB);   // [128][1024] = [x_t | h_prev]
    float* cbuf = (float*)(ws + OFF_C);
    float* Gi   = (float*)(ws + OFF_GI);           // Gi/Gu/Go contiguous
    float* fvec = (float*)(ws + OFF_F);
    const _Float16* Wt = (const _Float16*)(ws + OFF_WT);
    int* cnt = (int*)(ws + OFF_SYNC);
    int* gen = cnt + 1;

    const int lhalf = lane >> 4;        // 0 or 1
    const int lmod  = lane & 15;
    const int koff  = lhalf * 8;        // A-layout: lane half selects K+8

    for (int t = 0; t < TT; ++t) {
        // soft prefetch of next step's inputs into GL2 (global_prefetch_b8)
        if (t + 1 < TT)
            __builtin_prefetch(X + ((size_t)(t + 1) * BB * DD) + (size_t)tid * 64, 0, 1);

        // ---------------- Phase A: fused gate GEMM via WMMA ----------------
        // D(128x1536) = combined(128x1024) @ Wt(1024x1536)
        for (int tile = gw; tile < TOT_TILES; tile += NWAVES) {
            const int mt = tile / NTILES;           // 0..7
            const int nt = tile % NTILES;           // 0..95
            const int gate = nt >> 5;               // 0:i 1:u 2:o
            const int ntl = nt & 31;                // within-gate tile column
            const int n0 = ntl * 16;

            const int m = mt * 16 + lmod;           // A row for this lane
            const _Float16* arow = comb + (size_t)m * KDIM;
            v8f acc = {};

            #pragma unroll 4
            for (int kk = 0; kk < KSTEPS; ++kk) {
                const int k0 = kk * 32;
                // ---- A fragment (f16, pre-staged), doc layout:
                //   elems 0..7 -> K = k0+koff+0..7 ; elems 8..15 -> K+16
                v8h lo = *(const v8h*)(arow + k0 + koff);
                v8h hi = *(const v8h*)(arow + k0 + koff + 16);
                v16h A;
                #pragma unroll
                for (int j = 0; j < 8; ++j) { A[j] = lo[j]; A[j + 8] = hi[j]; }

                // ---- B fragment: lane holds column n0+lmod; elems 0..7 -> K = k0+koff+0..7 ; elems 8..15 -> K+16
                const _Float16* pb =
                    Wt + ((size_t)gate * KDIM + (size_t)(k0 + koff)) * HH + n0 + lmod;
                v16h Bf;
                #pragma unroll
                for (int j = 0; j < 8; ++j) { Bf[j] = pb[(size_t)j * HH]; Bf[j + 8] = pb[(size_t)(j + 16) * HH]; }

                acc = __builtin_amdgcn_wmma_f32_16x16x32_f16(
                    false, A, false, Bf, (short)0, acc, false, false);
            }

            // ---- epilogue: bias + branchless activation (gate is wave-uniform)
            //   sigmoid: s(x);   tanh: 2*s(2x)-1
            const float* bv = (gate == 0) ? bi : (gate == 1) ? bu : bo;
            const float bias  = bv[n0 + lmod];
            const float scale = (gate == 1) ? 2.0f : 1.0f;
            const float amul  = (gate == 1) ? 2.0f : 1.0f;
            const float aadd  = (gate == 1) ? -1.0f : 0.0f;
            // store in WMMA C-layout order -> fully coalesced 128B wave stores
            float* G = Gi + (size_t)gate * (BB * HH) + ((mt * 32 + ntl) * 256);
            #pragma unroll
            for (int v = 0; v < 8; ++v) {
                const float s = fast_sigmoid(scale * (acc[v] + bias));
                G[v * 32 + lane] = fmaf(amul, s, aadd);
            }
        }

        // ------- forget-gate MLP: blocks 0-1, 4 lanes per batch row ---------
        if (blockIdx.x < 2) {
            const int b = blockIdx.x * 64 + (tid >> 2);   // 0..127
            const int q = tid & 3;                        // k-quarter
            float l0 = 0.0f, l1 = 0.0f;
            const _Float16* cr = comb + (size_t)b * KDIM + q * (KDIM / 4);
            for (int k = 0; k < KDIM / 4; ++k) {
                const float v = (float)cr[k];
                const int kg = q * (KDIM / 4) + k;
                l0 = fmaf(v, Wf[kg], l0);
                l1 = fmaf(v, Wf[KDIM + kg], l1);
            }
            // cross-lane reduce over the 4-lane group (wave32 lane permutes)
            l0 += __shfl_xor(l0, 1, 32); l0 += __shfl_xor(l0, 2, 32);
            l1 += __shfl_xor(l1, 1, 32); l1 += __shfl_xor(l1, 2, 32);
            if (q == 0) {
                l0 += bfv[0]; l1 += bfv[1];
                float hdn[4];
                #pragma unroll
                for (int j = 0; j < 4; ++j)
                    hdn[j] = fast_tanh(fmaf(Ws1[j * 2], l0,
                                       fmaf(Ws1[j * 2 + 1], l1, bs1[j])));
                float s0 = bs2[0], s1 = bs2[1];
                #pragma unroll
                for (int j = 0; j < 4; ++j) {
                    s0 = fmaf(Ws2[j], hdn[j], s0);
                    s1 = fmaf(Ws2[4 + j], hdn[j], s1);
                }
                const float mx = fmaxf(s0, s1);
                const float e0 = __expf(s0 - mx), e1 = __expf(s1 - mx);
                fvec[b] = e0 * __builtin_amdgcn_rcpf(e0 + e1);
            }
        }

        grid_barrier(cnt, gen);

        // ---------------- Phase B: state update + stage next inputs ----------
        // iterate in C-layout order: coalesced gate reads + c read/write
        const int base = blockIdx.x * TPB + tid;
        for (int idx = base; idx < BB * HH; idx += NBLK * TPB) {
            const int tile = idx >> 8;              // (mt*32 + ntl)
            const int r    = idx & 255;
            const int v    = r >> 5;
            const int ln   = r & 31;
            const int b    = ((tile >> 5) << 4) + v + ((ln >> 4) << 3);
            const int j    = ((tile & 31) << 4) + (ln & 15);
            const float fb = fvec[b];
            const float cn = fmaf(fb, cbuf[idx],
                                  Gi[idx] * Gi[(size_t)(BB * HH) + idx]);
            const float hn = Gi[(size_t)2 * (BB * HH) + idx] * fast_tanh(cn);
            cbuf[idx] = cn;
            comb[(size_t)b * KDIM + DD + j] = (_Float16)hn;   // h_next (f16)
            const int e = b * HH + j;
            out[(size_t)t * (BB * HH) + e] = hn;
            if (t == TT - 1) {
                out[(size_t)TT * (BB * HH) + e]           = hn;  // final h
                out[(size_t)TT * (BB * HH) + BB * HH + e] = cn;  // final c
            }
        }
        // stage x_{t+1} into combined (f16) — consumed only after the barrier
        if (t + 1 < TT) {
            const float* xn = X + (size_t)(t + 1) * BB * DD;
            for (int e = base; e < BB * DD; e += NBLK * TPB) {
                const int b = e >> 9;
                const int j = e & (DD - 1);
                comb[(size_t)b * KDIM + j] = (_Float16)xn[e];
            }
        }

        grid_barrier(cnt, gen);   // combined/c visible before next step's GEMM
    }
}

extern "C" void kernel_launch(void* const* d_in, const int* in_sizes, int n_in,
                              void* d_out, int out_size, void* d_ws, size_t ws_size,
                              hipStream_t stream) {
    const float* X   = (const float*)d_in[0];   // inputs [T][B][D]
    const float* Wf  = (const float*)d_in[1];
    const float* bfv = (const float*)d_in[2];
    const float* Wi  = (const float*)d_in[3];
    const float* bi  = (const float*)d_in[4];
    const float* Wu  = (const float*)d_in[5];
    const float* bu  = (const float*)d_in[6];
    const float* Wo  = (const float*)d_in[7];
    const float* bo  = (const float*)d_in[8];
    const float* Ws1 = (const float*)d_in[9];
    const float* bs1 = (const float*)d_in[10];
    const float* Ws2 = (const float*)d_in[11];
    const float* bs2 = (const float*)d_in[12];
    char* ws   = (char*)d_ws;
    float* out = (float*)d_out;

    qlstm_init_convert<<<256, TPB, 0, stream>>>(X, Wi, Wu, Wo, ws);
    qlstm_persistent<<<NBLK, TPB, 0, stream>>>(X, Wf, bfv, bi, bu, bo,
                                               Ws1, bs1, Ws2, bs2, ws, out);
}
